// MambaClassifier_36447092473920
// MI455X (gfx1250) — matrix operand order repaired and checked by reading, not compiled, not run on hardware
//
#include <hip/hip_runtime.h>
#include <hip/hip_bf16.h>
#include <math.h>
#include <stdint.h>

// ---------------------------------------------------------------------------
// MambaClassifier forward for MI455X (gfx1250, wave32, WMMA, async-to-LDS)
// ---------------------------------------------------------------------------
#define B_SZ    8
#define L_SZ    4096
#define IN_SZ   1024
#define D_MODEL 256
#define D_INNER 512
#define D_STATE 16
#define D_CONV  4
#define DT_RANK 16
#define M_TOT   (B_SZ * L_SZ)      // 32768 rows

typedef __attribute__((ext_vector_type(16))) __bf16 v16bf;
typedef __attribute__((ext_vector_type(8)))  __bf16 v8bf;
typedef __attribute__((ext_vector_type(4)))  __bf16 v4bf;
typedef __attribute__((ext_vector_type(8)))  float  v8f;
typedef __attribute__((ext_vector_type(2)))  float  v2f;

// ---------------------------------------------------------------------------
// fp32 -> bf16 bulk converter (vectorized x4). n must be a multiple of 4.
// ---------------------------------------------------------------------------
__global__ void cvt_f32_bf16_kernel(const float* __restrict__ src,
                                    __bf16* __restrict__ dst, long n4) {
  long i = (long)blockIdx.x * blockDim.x + threadIdx.x;
  if (i >= n4) return;
  float4 v = *(const float4*)(src + i * 4);
  v4bf o;
  o[0] = (__bf16)v.x; o[1] = (__bf16)v.y; o[2] = (__bf16)v.z; o[3] = (__bf16)v.w;
  *(v4bf*)(dst + i * 4) = o;
}

// ---------------------------------------------------------------------------
// Epilogues
// ---------------------------------------------------------------------------
struct EpiFc1 {              // bias + exact GELU, store bf16
  const float* bias; __bf16* out;
  __device__ void operator()(int m, int n, float v) const {
    v += bias[n];
    v = 0.5f * v * (1.0f + erff(v * 0.70710678118654752f));
    out[(size_t)m * D_MODEL + n] = (__bf16)v;
  }
};
struct EpiInProj {           // split u / silu(z), store fp32
  float* u; float* zs;
  __device__ void operator()(int m, int n, float v) const {
    if (n < D_INNER) u[(size_t)m * D_INNER + n] = v;
    else             zs[(size_t)m * D_INNER + (n - D_INNER)] = v / (1.0f + __expf(-v));
  }
};
struct EpiXProj {
  float* out;
  __device__ void operator()(int m, int n, float v) const {
    out[(size_t)m * 48 + n] = v;
  }
};
struct EpiOutProj {
  float* out;
  __device__ void operator()(int m, int n, float v) const {
    out[(size_t)m * D_MODEL + n] = v;
  }
};

// ---------------------------------------------------------------------------
// bf16 WMMA GEMM: out[M,N] = A[M,K] * W[N,K]^T (+epilogue).
// A and W are bf16 (pre-converted). Block = 4 waves; each wave owns one
// 16-row M tile and register-blocks NTILES x 16 columns.
// The B tile (NTILES*16 rows x 32 k) is staged into LDS with the gfx1250
// async-DMA path (global_load_async_to_lds_b128, ASYNCcnt) and shared by
// all four waves; fragments are then read back via ds_load_b128.
//
// Fragment layouts (CDNA5 ISA):
//   A: lane l -> row m=l&15; e<8: k=k0+half*8+e ; e>=8: k=k0+16+half*8+e-8
//   B: lane l -> col n=l&15; e<8: k=k0+half*8+e ; e>=8: k=k0+16+half*8+e-8
//   C: lane l, reg r -> (m = r + 8*half, n = l&15)
// Requires M % 64 == 0, K % 32 == 0, N % (16*NTILES) == 0 (all hold here).
// ---------------------------------------------------------------------------
template <int NTILES, typename Epi>
__global__ __launch_bounds__(128) void gemm_bf16_kernel(
    const __bf16* __restrict__ Ab, int lda,
    const __bf16* __restrict__ W, int M, int N, int K, Epi epi) {
  __shared__ __bf16 Btile[NTILES * 16][32];
  const int wave = threadIdx.x >> 5;
  const int lane = threadIdx.x & 31;
  const int half = lane >> 4;
  const int nl   = lane & 15;
  const int m0   = (blockIdx.x * 4 + wave) * 16;

  for (int n0 = 0; n0 < N; n0 += 16 * NTILES) {
    v8f acc[NTILES] = {};
    for (int k0 = 0; k0 < K; k0 += 32) {
      __syncthreads();   // previous tile fully consumed before overwrite
      // --- async stage of B tile: rows n0..n0+16*NTILES-1, k k0..k0+31 ---
      for (int c = threadIdx.x; c < NTILES * 16 * 4; c += 128) {
        int n = c >> 2, part = c & 3;                       // 16B chunk
        unsigned lds_off = (unsigned)(uintptr_t)(&Btile[n][part * 8]);
        int g_off = (int)(((long)(n0 + n) * K + k0) * 2 + part * 16);
        asm volatile("global_load_async_to_lds_b128 %0, %1, %2"
                     :: "v"(lds_off), "v"(g_off), "s"(W) : "memory");
      }
      asm volatile("s_wait_asynccnt 0" ::: "memory");
      __syncthreads();

      // --- A fragment: two 16B loads of contiguous bf16 ---
      const __bf16* qa = Ab + (size_t)(m0 + nl) * lda + k0 + half * 8;
      if (k0 + 32 < K) __builtin_prefetch(qa + 32, 0, 0);
      v8bf alo = *(const v8bf*)(qa);
      v8bf ahi = *(const v8bf*)(qa + 16);
      v16bf a;
#pragma unroll
      for (int e = 0; e < 8; ++e) { a[e] = alo[e]; a[e + 8] = ahi[e]; }

#pragma unroll
      for (int t = 0; t < NTILES; ++t) {
        const __bf16* qb = &Btile[t * 16 + nl][half * 8];
        v8bf blo = *(const v8bf*)(qb);
        v8bf bhi = *(const v8bf*)(qb + 16);
        v16bf b;
#pragma unroll
        for (int e = 0; e < 8; ++e) { b[e] = blo[e]; b[e + 8] = bhi[e]; }
        acc[t] = __builtin_amdgcn_wmma_f32_16x16x32_bf16(
            false, a, false, b, (short)0, acc[t], false, false);
      }
    }
#pragma unroll
    for (int t = 0; t < NTILES; ++t)
#pragma unroll
      for (int r = 0; r < 8; ++r)
        epi(m0 + half * 8 + r, n0 + t * 16 + nl, acc[t][r]);
  }
}

// ---------------------------------------------------------------------------
// dt_proj: [M,16] x [512,16]^T, K=16 via 4 steps of f32 WMMA 16x16x4,
// epilogue = softplus(v + bias) using hardware exp/log.
// ---------------------------------------------------------------------------
__global__ __launch_bounds__(128) void dtproj_kernel(const float* __restrict__ xdbl,
                                                     const float* __restrict__ W,
                                                     const float* __restrict__ bias,
                                                     float* __restrict__ dtout, int M) {
  const int wave = threadIdx.x >> 5;
  const int lane = threadIdx.x & 31;
  const int half = lane >> 4;
  const int nl   = lane & 15;
  const int m0   = (blockIdx.x * 4 + wave) * 16;
  if (m0 >= M) return;

  for (int n0 = 0; n0 < D_INNER; n0 += 64) {
    v8f acc[4] = {};
#pragma unroll
    for (int k0 = 0; k0 < DT_RANK; k0 += 4) {
      const float* qa = xdbl + (size_t)(m0 + nl) * 48 + k0 + half * 2;
      v2f a; a[0] = qa[0]; a[1] = qa[1];
#pragma unroll
      for (int t = 0; t < 4; ++t) {
        const float* qb = W + (size_t)(n0 + t * 16 + nl) * DT_RANK + k0 + half * 2;
        v2f b; b[0] = qb[0]; b[1] = qb[1];
        acc[t] = __builtin_amdgcn_wmma_f32_16x16x4_f32(
            false, a, false, b, (short)0, acc[t], false, false);
      }
    }
#pragma unroll
    for (int t = 0; t < 4; ++t)
#pragma unroll
      for (int r = 0; r < 8; ++r) {
        int m = m0 + half * 8 + r, n = n0 + t * 16 + nl;
        float v = acc[t][r] + bias[n];
        dtout[(size_t)m * D_INNER + n] =
            (v > 20.0f) ? v : __logf(1.0f + __expf(v));
      }
  }
}

// ---------------------------------------------------------------------------
// Depthwise causal conv (k=4) + SiLU; writes fp32 (for scan) and bf16
// (for the x_proj GEMM A operand).
// ---------------------------------------------------------------------------
__global__ void conv_silu_kernel(const float* __restrict__ u,
                                 const float* __restrict__ cw,
                                 const float* __restrict__ cb,
                                 float* __restrict__ uc,
                                 __bf16* __restrict__ ucb) {
  size_t i  = (size_t)blockIdx.x * blockDim.x + threadIdx.x;
  int    d  = (int)(i % D_INNER);
  size_t bl = i / D_INNER;              // b*L + t
  int    t  = (int)(bl % L_SZ);
  float acc = cb[d];
#pragma unroll
  for (int j = 0; j < D_CONV; ++j) {
    int tt = t - (D_CONV - 1) + j;
    if (tt >= 0)
      acc += cw[d * D_CONV + j] * u[(bl - (size_t)(D_CONV - 1) + j) * D_INNER + d];
  }
  float s = acc / (1.0f + __expf(-acc));  // silu
  uc[i]  = s;
  ucb[i] = (__bf16)s;
}

// ---------------------------------------------------------------------------
// Selective scan: one thread per (b,d) chain; 16-float state in VGPRs;
// B/C vectors staged per 64-step chunk in LDS. Epilogue fuses
// (y + u*D) * silu(z) and writes bf16 for out_proj.
// ---------------------------------------------------------------------------
#define SCAN_CHUNK 64
__global__ __launch_bounds__(128) void scan_kernel(const float* __restrict__ dt,
                                                   const float* __restrict__ uc,
                                                   const float* __restrict__ xdbl,
                                                   const float* __restrict__ A_log,
                                                   const float* __restrict__ D_skip,
                                                   const float* __restrict__ zs,
                                                   __bf16* __restrict__ ybf) {
  const int b = blockIdx.x >> 2;                   // 8 batches x 4 d-blocks
  const int d = (blockIdx.x & 3) * 128 + threadIdx.x;

  float Av[D_STATE];
#pragma unroll
  for (int s = 0; s < D_STATE; ++s) Av[s] = -__expf(A_log[d * D_STATE + s]);
  const float Dv = D_skip[d];

  float st[D_STATE] = {};
  __shared__ float Bs[SCAN_CHUNK][D_STATE];
  __shared__ float Cs[SCAN_CHUNK][D_STATE];

  for (int t0 = 0; t0 < L_SZ; t0 += SCAN_CHUNK) {
    for (int i = threadIdx.x; i < SCAN_CHUNK * 32; i += 128) {
      int tt = i >> 5, c = i & 31;
      float v = xdbl[((size_t)(b * L_SZ + t0 + tt)) * 48 + DT_RANK + c];
      if (c < D_STATE) Bs[tt][c] = v;
      else             Cs[tt][c - D_STATE] = v;
    }
    __syncthreads();
    for (int tt = 0; tt < SCAN_CHUNK; ++tt) {
      size_t md  = ((size_t)(b * L_SZ + t0 + tt)) * D_INNER + d;
      float dtv = dt[md];
      float uv  = uc[md];
      float dtu = dtv * uv;
      float y   = 0.0f;
#pragma unroll
      for (int s = 0; s < D_STATE; ++s) {
        float dA = __expf(dtv * Av[s]);
        st[s] = dA * st[s] + dtu * Bs[tt][s];
        y += st[s] * Cs[tt][s];
      }
      ybf[md] = (__bf16)((y + uv * Dv) * zs[md]);
    }
    __syncthreads();
  }
}

// ---------------------------------------------------------------------------
// Mean pool over L, then tiny fc3.
// ---------------------------------------------------------------------------
__global__ void pool_kernel(const float* __restrict__ h, float* __restrict__ pooled) {
  int b = blockIdx.x, n = threadIdx.x;   // 8 x 256
  float s = 0.0f;
  for (int t = 0; t < L_SZ; ++t) s += h[((size_t)(b * L_SZ + t)) * D_MODEL + n];
  pooled[b * D_MODEL + n] = s * (1.0f / (float)L_SZ);
}

__global__ void fc3_kernel(const float* __restrict__ pooled,
                           const float* __restrict__ w,
                           const float* __restrict__ bias,
                           float* __restrict__ out) {
  int i = threadIdx.x;
  if (i >= B_SZ * 2) return;
  int b = i >> 1, c = i & 1;
  float s = bias[c];
  for (int k = 0; k < D_MODEL; ++k) s += pooled[b * D_MODEL + k] * w[c * D_MODEL + k];
  out[b * 2 + c] = s;
}

// ---------------------------------------------------------------------------
// Host-side orchestration
// ---------------------------------------------------------------------------
extern "C" void kernel_launch(void* const* d_in, const int* in_sizes, int n_in,
                              void* d_out, int out_size, void* d_ws, size_t ws_size,
                              hipStream_t stream) {
  (void)in_sizes; (void)n_in; (void)out_size; (void)ws_size;
  const float* x          = (const float*)d_in[0];
  const float* fc1_w      = (const float*)d_in[1];
  const float* fc1_b      = (const float*)d_in[2];
  const float* in_proj_w  = (const float*)d_in[3];
  const float* conv_w     = (const float*)d_in[4];
  const float* conv_b     = (const float*)d_in[5];
  const float* x_proj_w   = (const float*)d_in[6];
  const float* dt_proj_w  = (const float*)d_in[7];
  const float* dt_proj_b  = (const float*)d_in[8];
  const float* A_log      = (const float*)d_in[9];
  const float* D_skip     = (const float*)d_in[10];
  const float* out_proj_w = (const float*)d_in[11];
  const float* fc3_w      = (const float*)d_in[12];
  const float* fc3_b      = (const float*)d_in[13];
  float* out = (float*)d_out;

  // Workspace layout. Region r1 is reused (u_pre -> dt -> out_proj output);
  // region xreg is reused (xb -> {ucb, ybf}).
  char*   w8     = (char*)d_ws;
  float*  r1     = (float*) (w8 + 0);                    // 64 MB
  float*  zs     = (float*) (w8 + (size_t)67108864);     // 64 MB
  float*  uc     = (float*) (w8 + (size_t)134217728);    // 64 MB
  char*   xreg   =          (w8 + (size_t)201326592);    // 64 MB (reused)
  __bf16* xb     = (__bf16*)(xreg);                      // bf16 copy of x
  __bf16* ucb    = (__bf16*)(xreg);                      // after fc1
  __bf16* ybf    = (__bf16*)(xreg + (size_t)33554432);   // after x_proj
  __bf16* hb     = (__bf16*)(w8 + (size_t)268435456);    // 16 MB
  float*  xdbl   = (float*) (w8 + (size_t)285212672);    // 6 MB
  __bf16* fc1_wb = (__bf16*)(w8 + (size_t)291504128);
  __bf16* inp_wb = (__bf16*)(w8 + (size_t)292028416);
  __bf16* xp_wb  = (__bf16*)(w8 + (size_t)292552704);
  __bf16* op_wb  = (__bf16*)(w8 + (size_t)292601856);
  float*  pooled = (float*) (w8 + (size_t)292864000);

  const int gemm_blocks = M_TOT / 64;  // 4 waves x 16 rows per block

  // 0) pre-convert GEMM operands to bf16 (one pass each)
  cvt_f32_bf16_kernel<<<(M_TOT * IN_SZ / 4 + 255) / 256, 256, 0, stream>>>(
      x, xb, (long)M_TOT * IN_SZ / 4);
  cvt_f32_bf16_kernel<<<(D_MODEL * IN_SZ / 4 + 255) / 256, 256, 0, stream>>>(
      fc1_w, fc1_wb, (long)D_MODEL * IN_SZ / 4);
  cvt_f32_bf16_kernel<<<(2 * D_INNER * D_MODEL / 4 + 255) / 256, 256, 0, stream>>>(
      in_proj_w, inp_wb, (long)2 * D_INNER * D_MODEL / 4);
  cvt_f32_bf16_kernel<<<(48 * D_INNER / 4 + 255) / 256, 256, 0, stream>>>(
      x_proj_w, xp_wb, (long)48 * D_INNER / 4);
  cvt_f32_bf16_kernel<<<(D_MODEL * D_INNER / 4 + 255) / 256, 256, 0, stream>>>(
      out_proj_w, op_wb, (long)D_MODEL * D_INNER / 4);

  // 1) fc1 + GELU -> h (bf16)
  gemm_bf16_kernel<4><<<gemm_blocks, 128, 0, stream>>>(
      xb, IN_SZ, fc1_wb, M_TOT, D_MODEL, IN_SZ, EpiFc1{fc1_b, hb});

  // 2) in_proj -> u_pre (fp32) + silu(z) (fp32)
  gemm_bf16_kernel<4><<<gemm_blocks, 128, 0, stream>>>(
      hb, D_MODEL, inp_wb, M_TOT, 2 * D_INNER, D_MODEL, EpiInProj{r1, zs});

  // 3) depthwise causal conv + silu -> uc (fp32) + ucb (bf16)
  conv_silu_kernel<<<(M_TOT * D_INNER) / 256, 256, 0, stream>>>(
      r1, conv_w, conv_b, uc, ucb);

  // 4) x_proj -> x_dbl [M,48]
  gemm_bf16_kernel<3><<<gemm_blocks, 128, 0, stream>>>(
      ucb, D_INNER, xp_wb, M_TOT, 48, D_INNER, EpiXProj{xdbl});

  // 5) dt_proj (+softplus) -> dt (reuses r1)
  dtproj_kernel<<<gemm_blocks, 128, 0, stream>>>(xdbl, dt_proj_w, dt_proj_b, r1, M_TOT);

  // 6) selective scan, fused (y + u*D)*silu(z) -> ybf (bf16)
  scan_kernel<<<B_SZ * 4, 128, 0, stream>>>(r1, uc, xdbl, A_log, D_skip, zs, ybf);

  // 7) out_proj -> r1 (fp32, reuse)
  gemm_bf16_kernel<4><<<gemm_blocks, 128, 0, stream>>>(
      ybf, D_INNER, op_wb, M_TOT, D_MODEL, D_INNER, EpiOutProj{r1});

  // 8) mean pool over L, 9) fc3
  pool_kernel<<<B_SZ, D_MODEL, 0, stream>>>(r1, pooled);
  fc3_kernel<<<1, 32, 0, stream>>>(pooled, fc3_w, fc3_b, out);
}
